// SRUCell_85444079386966
// MI455X (gfx1250) — matrix operand order repaired and checked by reading, not compiled, not run on hardware
//
#include <hip/hip_runtime.h>
#include <cmath>
#include <cstdint>
#include <cstddef>

typedef __bf16 bf16_t;
typedef bf16_t v16bf __attribute__((ext_vector_type(16)));
typedef bf16_t v4bf  __attribute__((ext_vector_type(4)));
typedef float  v8f   __attribute__((ext_vector_type(8)));
typedef unsigned int u32x4 __attribute__((ext_vector_type(4)));

#define T_DIM 2048
#define B_DIM 16
#define D_DIM 1024
#define N_DIM 3072                 // 3*D
#define M_DIM (T_DIM * B_DIM)      // 32768
#define K_DIM 1024

// ---------- fast device math (guarded so the host pass still parses) ----------
__device__ __forceinline__ float fast_rcp(float x) {
#if defined(__HIP_DEVICE_COMPILE__)
  return __builtin_amdgcn_rcpf(x);
#else
  return 1.0f / x;
#endif
}
__device__ __forceinline__ float fast_exp(float x) {
#if defined(__HIP_DEVICE_COMPILE__)
  return __expf(x);
#else
  return expf(x);
#endif
}
__device__ __forceinline__ float fast_sigmoid(float x) {
  return fast_rcp(1.0f + fast_exp(-x));
}
__device__ __forceinline__ float fast_tanh(float x) {
#if defined(__HIP_DEVICE_COMPILE__) && __has_builtin(__builtin_amdgcn_tanhf)
  return __builtin_amdgcn_tanhf(x);            // v_tanh_f32 on gfx1250
#elif defined(__HIP_DEVICE_COMPILE__)
  float a = fabsf(x);
  float e = fast_exp(-2.0f * a);
  float t = (1.0f - e) * fast_rcp(1.0f + e);
  return copysignf(t, x);
#else
  return tanhf(x);
#endif
}

// ---------- WMMA fragment composition: two 16B LDS chunks -> v16bf ----------
struct Frag2 { u32x4 a, b; };
__device__ __forceinline__ v16bf frag_from(const void* p0, const void* p1) {
  Frag2 f;
  f.a = *(const u32x4*)p0;
  f.b = *(const u32x4*)p1;
  return __builtin_bit_cast(v16bf, f);
}

// =====================================================================
// Kernel 0: W [K, N] fp32  ->  transposed bf16 hi/lo planes [N, K] in ws
// =====================================================================
__global__ __launch_bounds__(256) void wconv_kernel(const float* __restrict__ W,
                                                    bf16_t* __restrict__ Wth,
                                                    bf16_t* __restrict__ Wtl) {
  int idx = blockIdx.x * 256 + threadIdx.x;   // 0 .. N*K/4-1
  int n   = idx % N_DIM;                      // coalesced across lanes
  int kg  = idx / N_DIM;                      // 0..255
  int k0  = kg * 4;
  v4bf h, l;
#pragma unroll
  for (int j = 0; j < 4; ++j) {
    float w    = W[(size_t)(k0 + j) * N_DIM + n];
    bf16_t hh  = (bf16_t)w;
    h[j] = hh;
    l[j] = (bf16_t)(w - (float)hh);
  }
  *(v4bf*)(Wth + (size_t)n * K_DIM + k0) = h;
  *(v4bf*)(Wtl + (size_t)n * K_DIM + k0) = l;
}

// =====================================================================
// Kernel 1: U[M,N] = X[M,K] @ W[K,N] via bf16x3 WMMA, fused bias+sigmoid
// Block tile 128x128, 8 waves of 32x64, K-step 32.
// =====================================================================
__global__ __launch_bounds__(256) void sru_gemm_kernel(
    const float*  __restrict__ X,     // [M, K] fp32
    const bf16_t* __restrict__ Wth,   // [N, K] bf16 hi
    const bf16_t* __restrict__ Wtl,   // [N, K] bf16 lo
    const float*  __restrict__ bias,  // [2*D]
    float*        __restrict__ U)     // [M, N] activated gates
{
  __shared__ bf16_t sAh[128 * 32];
  __shared__ bf16_t sAl[128 * 32];
  __shared__ bf16_t sBh[128 * 32];
  __shared__ bf16_t sBl[128 * 32];

  const int tid  = threadIdx.x;
  const int lane = tid & 31;
  const int wave = tid >> 5;       // 0..7
  const int wm   = wave >> 1;      // 0..3  (M strip of 32)
  const int wn   = wave & 1;       // 0..1  (N strip of 64)
  const int m0   = blockIdx.y * 128;
  const int n0   = blockIdx.x * 128;

  // staging thread maps
  const int arow   = tid >> 3;          // 0..31 (A: 8 threads x float4 per row)
  const int acol   = (tid & 7) * 4;     // element col in 32
  const int brow   = tid >> 2;          // 0..63 (B: 4 threads x 16B per 64B row)
  const int bcol16 = (tid & 3) * 16;    // byte offset

  float4 aReg[4];
  u32x4  bhReg[2], blReg[2];

  auto gload = [&](int kt) {
    const int kk = kt * 32;
#pragma unroll
    for (int p = 0; p < 4; ++p) {
      const int m = m0 + arow + 32 * p;
      aReg[p] = *(const float4*)(X + (size_t)m * K_DIM + kk + acol);
    }
#pragma unroll
    for (int p = 0; p < 2; ++p) {
      const int n = n0 + brow + 64 * p;
      const char* ph = (const char*)Wth + ((size_t)n * K_DIM + kk) * 2 + bcol16;
      const char* pl = (const char*)Wtl + ((size_t)n * K_DIM + kk) * 2 + bcol16;
      bhReg[p] = *(const u32x4*)ph;
      blReg[p] = *(const u32x4*)pl;
    }
  };

  auto stage = [&]() {
#pragma unroll
    for (int p = 0; p < 4; ++p) {
      float vv[4] = {aReg[p].x, aReg[p].y, aReg[p].z, aReg[p].w};
      v4bf h, l;
#pragma unroll
      for (int j = 0; j < 4; ++j) {
        bf16_t hh = (bf16_t)vv[j];
        h[j] = hh;
        l[j] = (bf16_t)(vv[j] - (float)hh);
      }
      const int row = arow + 32 * p;
      *(v4bf*)(sAh + row * 32 + acol) = h;
      *(v4bf*)(sAl + row * 32 + acol) = l;
    }
#pragma unroll
    for (int p = 0; p < 2; ++p) {
      const int row = brow + 64 * p;
      *(u32x4*)((char*)sBh + row * 64 + bcol16) = bhReg[p];
      *(u32x4*)((char*)sBl + row * 64 + bcol16) = blReg[p];
    }
  };

  v8f acc[2][4];
#pragma unroll
  for (int i = 0; i < 2; ++i)
#pragma unroll
    for (int j = 0; j < 4; ++j)
      acc[i][j] = (v8f){0.f, 0.f, 0.f, 0.f, 0.f, 0.f, 0.f, 0.f};

  // per-lane fragment offsets
  // A 16x32 bf16 (row m = lane&15): lane<16 -> K{0..7,16..23}, lane>=16 -> +8
  const int aRow0 = wm * 32 + (lane & 15);
  const int aHalf = (lane >> 4) * 16;   // byte offset of first 16B run in 64B row
  // B 32x16 bf16 (col n = lane&15): lane<16 -> K{0..7,16..23}, lane>=16 -> +8
  const int bRow0 = wn * 64 + (lane & 15);
  const int bHalf = (lane >> 4) * 16;   // byte offset of first 16B run in 64B row

  gload(0);
  for (int kt = 0; kt < 32; ++kt) {
    stage();
    __syncthreads();
    if (kt + 1 < 32) gload(kt + 1);     // prefetch next K-slice behind compute

    v16bf ah[2], al[2], bh[4], bl[4];
#pragma unroll
    for (int sm = 0; sm < 2; ++sm) {
      const char* bh_ = (const char*)sAh + (aRow0 + sm * 16) * 64 + aHalf;
      const char* bl_ = (const char*)sAl + (aRow0 + sm * 16) * 64 + aHalf;
      ah[sm] = frag_from(bh_, bh_ + 32);
      al[sm] = frag_from(bl_, bl_ + 32);
    }
#pragma unroll
    for (int sn = 0; sn < 4; ++sn) {
      const char* bh_ = (const char*)sBh + (bRow0 + sn * 16) * 64 + bHalf;
      const char* bl_ = (const char*)sBl + (bRow0 + sn * 16) * 64 + bHalf;
      bh[sn] = frag_from(bh_, bh_ + 32);
      bl[sn] = frag_from(bl_, bl_ + 32);
    }
#pragma unroll
    for (int sm = 0; sm < 2; ++sm)
#pragma unroll
      for (int sn = 0; sn < 4; ++sn) {
        acc[sm][sn] = __builtin_amdgcn_wmma_f32_16x16x32_bf16(
            false, ah[sm], false, bh[sn], (short)0, acc[sm][sn], false, false);
        acc[sm][sn] = __builtin_amdgcn_wmma_f32_16x16x32_bf16(
            false, ah[sm], false, bl[sn], (short)0, acc[sm][sn], false, false);
        acc[sm][sn] = __builtin_amdgcn_wmma_f32_16x16x32_bf16(
            false, al[sm], false, bh[sn], (short)0, acc[sm][sn], false, false);
      }
    __syncthreads();
  }

  // epilogue: n = 3*d + g ; g==1 -> sigmoid(+forget_bias), g==2 -> sigmoid(+reset_bias)
  const int nl = lane & 15;
  const int mh = (lane >> 4) * 8;
#pragma unroll
  for (int sm = 0; sm < 2; ++sm) {
#pragma unroll
    for (int sn = 0; sn < 4; ++sn) {
      const int n = n0 + wn * 64 + sn * 16 + nl;
      const int d = n / 3;
      const int g = n - d * 3;
      float bb = 0.0f;
      if (g == 1)      bb = bias[d];
      else if (g == 2) bb = bias[D_DIM + d];
      const int mbase = m0 + wm * 32 + sm * 16 + mh;
#pragma unroll
      for (int i = 0; i < 8; ++i) {
        float v = acc[sm][sn][i];
        if (g != 0) v = fast_sigmoid(v + bb);
        U[(size_t)(mbase + i) * N_DIM + n] = v;
      }
    }
  }
}

// =====================================================================
// Kernel 2: sequential scan over T; one thread per (b, d) lane.
// c_t = x~ + f*(c - x~) ; h_t = x' + r*(tanh(c_t) - x')
// =====================================================================
__global__ __launch_bounds__(256) void sru_scan_kernel(
    const float* __restrict__ U,    // [T*B, 3D] activated gates
    const float* __restrict__ X,    // [T, B, D] (x')
    const float* __restrict__ C0,   // [B, D]
    float*       __restrict__ Out)  // h [T,B,D] then c_last [B,D]
{
  const int j = blockIdx.x * 256 + threadIdx.x;   // j = b*D + d, 0..16383
  float c = C0[j];
  const float* up = U + (size_t)j * 3;
  const float* xp = X + j;
  float*       hp = Out + j;
#pragma unroll 4
  for (int t = 0; t < T_DIM; ++t) {
    const float xt = up[0];
    const float f  = up[1];
    const float r  = up[2];
    const float xv = *xp;
    c = xt + f * (c - xt);
    const float th = fast_tanh(c);
    *hp = xv + r * (th - xv);
    up += (size_t)B_DIM * D_DIM * 3;
    xp += (size_t)B_DIM * D_DIM;
    hp += (size_t)B_DIM * D_DIM;
  }
  Out[(size_t)T_DIM * B_DIM * D_DIM + j] = c;
}

// =====================================================================
extern "C" void kernel_launch(void* const* d_in, const int* in_sizes, int n_in,
                              void* d_out, int out_size, void* d_ws, size_t ws_size,
                              hipStream_t stream) {
  (void)in_sizes; (void)n_in; (void)out_size; (void)ws_size;
  const float* X    = (const float*)d_in[0];   // [T, B, D]
  const float* C0   = (const float*)d_in[1];   // [B, D]
  const float* W    = (const float*)d_in[2];   // [D, 3D]
  const float* bias = (const float*)d_in[3];   // [2D]
  float* out = (float*)d_out;

  char* ws = (char*)d_ws;
  bf16_t* Wth = (bf16_t*)ws;
  bf16_t* Wtl = Wth + (size_t)N_DIM * K_DIM;
  float*  U   = (float*)(ws + 2 * (size_t)N_DIM * K_DIM * sizeof(bf16_t)); // +12 MB

  wconv_kernel<<<(N_DIM * (K_DIM / 4)) / 256, 256, 0, stream>>>(W, Wth, Wtl);

  dim3 grid(N_DIM / 128, M_DIM / 128);   // 24 x 256 = 6144 blocks
  sru_gemm_kernel<<<grid, 256, 0, stream>>>(X, Wth, Wtl, bias, U);

  sru_scan_kernel<<<(B_DIM * D_DIM) / 256, 256, 0, stream>>>(U, X, C0, out);
}
